// TransolverAttention_12481174962454
// MI455X (gfx1250) — compile-verified, matrix operand order checked by reading, not run on hardware
//
// Transolver attention for MI455X (gfx1250), wave32 + WMMA bf16.
// All heavy contractions (incl. slice-norm) via v_wmma_f32_16x16x32_bf16.
#include <hip/hip_runtime.h>
#include <hip/hip_bf16.h>
#include <stdint.h>

#define HH 8
#define CC 32
#define SS 64
#define DD 256

typedef __attribute__((ext_vector_type(16))) __bf16 v16bf;
typedef __attribute__((ext_vector_type(8)))  __bf16 v8bf;
typedef __attribute__((ext_vector_type(2)))  __bf16 v2bf;
typedef __attribute__((ext_vector_type(8)))  float  v8f;

static __device__ __forceinline__ __bf16 f2bf(float f) { return (__bf16)f; }
static __device__ __forceinline__ float bf2f(__bf16 b) { return (float)b; }
static __device__ __forceinline__ uint32_t pk2(float a, float b) {
  v2bf t; t[0] = (__bf16)a; t[1] = (__bf16)b;
  return __builtin_bit_cast(uint32_t, t);
}
static __device__ __forceinline__ v8f zero8() {
  v8f z = {0.f,0.f,0.f,0.f,0.f,0.f,0.f,0.f};
  return z;
}
static __device__ __forceinline__ v16bf combine(v8bf lo, v8bf hi) {
  return __builtin_shufflevector(lo, hi, 0,1,2,3,4,5,6,7,8,9,10,11,12,13,14,15);
}
static __device__ __forceinline__ v8f wmma_bf(v16bf a, v16bf b, v8f c) {
  return __builtin_amdgcn_wmma_f32_16x16x32_bf16(false, a, false, b, (short)0, c, false, false);
}

// A/B fragment from bf16 matrix stored row-major with row stride rs (elements).
// Per ISA 7.12.2: lanes 0-15 hold K = {k0..k0+7, k0+16..k0+23}, lanes 16-31 the +8 runs.
static __device__ __forceinline__ v16bf bfrag(const __bf16* base, int row, int rs, int k0, int lhalf) {
  const __bf16* p = base + (size_t)row * rs + k0 + (lhalf << 3);
  v8bf lo = *(const v8bf*)(p);
  v8bf hi = *(const v8bf*)(p + 16);
  return combine(lo, hi);
}

// A fragment straight from global fp32 row (x matrix), converted with native cvt.
static __device__ __forceinline__ v16bf afrag_f32(const float* rowp, int k0, int lhalf) {
  const float* p = rowp + k0 + (lhalf << 3);
  v8f lo = *(const v8f*)(p);
  v8f hi = *(const v8f*)(p + 16);
  return combine(__builtin_convertvector(lo, v8bf), __builtin_convertvector(hi, v8bf));
}

// ---------------- prep: transpose+convert weights to bf16 [n][k] ----------------
__global__ __launch_bounds__(256) void k_prep(
    const float* __restrict__ Wfx, const float* __restrict__ Wx,
    const float* __restrict__ Wp,  const float* __restrict__ Wsl,
    __bf16* __restrict__ wfxt, __bf16* __restrict__ wxt,
    __bf16* __restrict__ wpt,  __bf16* __restrict__ wslt) {
  int i = blockIdx.x * 256 + threadIdx.x;
  if (i < 65536)        { int k = i >> 8, n = i & 255;            wfxt[n * 256 + k] = f2bf(Wfx[i]); }
  else if (i < 131072)  { int j = i - 65536;  int k = j >> 8, n = j & 255; wxt[n * 256 + k] = f2bf(Wx[j]); }
  else if (i < 196608)  { int j = i - 131072; int k = j >> 8, n = j & 255; wpt[n * 256 + k] = f2bf(Wp[j]); }
  else if (i < 198656)  { int j = i - 196608; int c = j >> 6, s = j & 63;  wslt[s * 32 + c] = f2bf(Wsl[j]); }
}

// ---------------- pass1: x_mid/fx_mid GEMMs, softmax routing, pooled slice tokens ----------------
__global__ __launch_bounds__(256) void k_pass1(
    const float* __restrict__ x,
    const __bf16* __restrict__ wxt, const __bf16* __restrict__ wfxt,
    const __bf16* __restrict__ wslt,
    const float* __restrict__ bx, const float* __restrict__ bfx,
    const float* __restrict__ bsl, const float* __restrict__ temp,
    float* __restrict__ part, int nTok, int iters) {
  __shared__ __align__(16) __bf16 sWsl[SS * CC];
  __shared__ __align__(16) __bf16 sXm[8][32 * CC];   // [tok][c] per wave
  __shared__ __align__(16) __bf16 sFx[8][CC * 32];   // [c][tok] per wave
  __shared__ __align__(16) __bf16 sW [8][SS * 32];   // [s][tok] per wave
  __shared__ float sBx[DD], sBfx[DD], sBsl[SS], sInvT[HH];

  const int tid = threadIdx.x;
  const int wid = tid >> 5;       // wave id == head id
  const int lane = tid & 31;
  const int lhalf = lane >> 4;
  const int l15 = lane & 15;
  const int h = wid;

  if (tid < DD) { sBx[tid] = bx[tid]; sBfx[tid] = bfx[tid]; }
  if (tid < SS) sBsl[tid] = bsl[tid];
  if (tid < HH) sInvT[tid] = 1.0f / temp[tid];
  for (int i = tid; i < SS * CC; i += 256) sWsl[i] = wslt[i];
  __syncthreads();

  // ones-in-column-0 B fragment: norm[s] = sum_tok w[s][tok] via WMMA
  v16bf bones;
  {
    __bf16 o = (__bf16)((l15 == 0) ? 1.0f : 0.0f);
#pragma unroll
    for (int i = 0; i < 16; ++i) bones[i] = o;
  }

  v8f stacc[4][2];
  v8f nacc[4];
#pragma unroll
  for (int mt = 0; mt < 4; ++mt) {
    nacc[mt] = zero8();
#pragma unroll
    for (int nt = 0; nt < 2; ++nt) stacc[mt][nt] = zero8();
  }

  for (int it = 0; it < iters; ++it) {
    const int tb = (blockIdx.x * iters + it) * 32;
    if (tb >= nTok) break;
    __builtin_prefetch(x + (size_t)(tb + 32) * DD, 0, 1);

    // x_mid & fx_mid for this head's 32 output channels (K = 256)
    v8f xm[2][2], fxa[2][2];
#pragma unroll
    for (int mt = 0; mt < 2; ++mt)
#pragma unroll
      for (int nt = 0; nt < 2; ++nt) { xm[mt][nt] = zero8(); fxa[mt][nt] = zero8(); }

#pragma unroll
    for (int k = 0; k < 8; ++k) {
      v16bf a0 = afrag_f32(x + (size_t)(tb + l15) * DD,      k * 32, lhalf);
      v16bf a1 = afrag_f32(x + (size_t)(tb + 16 + l15) * DD, k * 32, lhalf);
#pragma unroll
      for (int nt = 0; nt < 2; ++nt) {
        const int n0 = h * CC + nt * 16 + l15;
        v16bf bxf = bfrag(wxt,  n0, DD, k * 32, lhalf);
        v16bf bff = bfrag(wfxt, n0, DD, k * 32, lhalf);
        xm[0][nt]  = wmma_bf(a0, bxf, xm[0][nt]);
        xm[1][nt]  = wmma_bf(a1, bxf, xm[1][nt]);
        fxa[0][nt] = wmma_bf(a0, bff, fxa[0][nt]);
        fxa[1][nt] = wmma_bf(a1, bff, fxa[1][nt]);
      }
    }

    // stage x_mid -> [tok][c] (+b_x), fx_mid -> [c][tok] (+b_fx)
#pragma unroll
    for (int mt = 0; mt < 2; ++mt)
#pragma unroll
      for (int nt = 0; nt < 2; ++nt) {
        const int c = nt * 16 + l15;
        const float bb = sBx[h * CC + c];
        __bf16* dst = &sXm[wid][(mt * 16 + 8 * lhalf) * CC + c];
#pragma unroll
        for (int r = 0; r < 8; ++r) dst[r * CC] = f2bf(xm[mt][nt][r] + bb);
        const float bf_ = sBfx[h * CC + c];
        __bf16* dst2 = &sFx[wid][c * 32 + mt * 16 + 8 * lhalf];
#pragma unroll
        for (int r = 0; r < 8; r += 2)
          *(uint32_t*)(dst2 + r) = pk2(fxa[mt][nt][r] + bf_, fxa[mt][nt][r + 1] + bf_);
      }

    // routing logits: [32 tok] x [K=32] @ W_slice -> stored [s][tok]
#pragma unroll
    for (int mt = 0; mt < 2; ++mt) {
      v16bf a = bfrag(sXm[wid], mt * 16 + l15, CC, 0, lhalf);
#pragma unroll
      for (int nt = 0; nt < 4; ++nt) {
        v16bf b = bfrag(sWsl, nt * 16 + l15, CC, 0, lhalf);
        v8f lg = wmma_bf(a, b, zero8());
        const int s = nt * 16 + l15;
        __bf16* dst = &sW[wid][s * 32 + mt * 16 + 8 * lhalf];
#pragma unroll
        for (int r = 0; r < 8; r += 2)
          *(uint32_t*)(dst + r) = pk2(lg[r], lg[r + 1]);
      }
    }

    // softmax over S for token = lane (3-pass over the [s][tok] tile)
    {
      const float invT = sInvT[h];
      float mx = -3.0e38f;
      for (int s = 0; s < SS; ++s) {
        float v = (bf2f(sW[wid][s * 32 + lane]) + sBsl[s]) * invT;
        mx = fmaxf(mx, v);
      }
      float sum = 0.f;
      for (int s = 0; s < SS; ++s) {
        float v = (bf2f(sW[wid][s * 32 + lane]) + sBsl[s]) * invT;
        float e = __expf(v - mx);
        sum += e;
        sW[wid][s * 32 + lane] = f2bf(e);
      }
      const float rn = 1.0f / sum;
      for (int s = 0; s < SS; ++s)
        sW[wid][s * 32 + lane] = f2bf(bf2f(sW[wid][s * 32 + lane]) * rn);
    }

    // pooled slice tokens: st[s][c] += w^T @ fx (M=64, K=32 tokens, N=32)
    // plus norm[s] += w^T @ ones (one extra WMMA per m-tile, column 0 of D)
#pragma unroll
    for (int mt = 0; mt < 4; ++mt) {
      v16bf a = bfrag(sW[wid], mt * 16 + l15, 32, 0, lhalf);
      nacc[mt] = wmma_bf(a, bones, nacc[mt]);
#pragma unroll
      for (int nt = 0; nt < 2; ++nt) {
        v16bf b = bfrag(sFx[wid], nt * 16 + l15, 32, 0, lhalf);
        stacc[mt][nt] = wmma_bf(a, b, stacc[mt][nt]);
      }
    }
  }

  // deterministic partial flush (no atomics, no barrier needed)
  float* pb = part + (size_t)blockIdx.x * 16896;
#pragma unroll
  for (int mt = 0; mt < 4; ++mt)
#pragma unroll
    for (int nt = 0; nt < 2; ++nt) {
      const int c = nt * 16 + l15;
#pragma unroll
      for (int r = 0; r < 8; ++r) {
        const int s = mt * 16 + r + 8 * lhalf;
        pb[h * (SS * CC) + s * CC + c] = stacc[mt][nt][r];
      }
    }
  if (l15 == 0) {
#pragma unroll
    for (int mt = 0; mt < 4; ++mt)
#pragma unroll
      for (int r = 0; r < 8; ++r) {
        const int s = mt * 16 + r + 8 * lhalf;
        pb[HH * SS * CC + h * SS + s] = nacc[mt][r];
      }
  }
}

// ---------------- reduce partials ----------------
__global__ __launch_bounds__(256) void k_reduce(const float* __restrict__ part,
                                                float* __restrict__ st,
                                                float* __restrict__ nrm, int nblk) {
  int idx = blockIdx.x * 256 + threadIdx.x;
  if (idx >= 16896) return;
  float s = 0.f;
  for (int b = 0; b < nblk; ++b) s += part[(size_t)b * 16896 + idx];
  if (idx < 16384) st[idx] = s;
  else nrm[idx - 16384] = s;
}

// ---------------- tiny slice SDPA (one block, one wave per head) ----------------
__global__ __launch_bounds__(256) void k_slice_attn(const float* __restrict__ st,
                                                    const float* __restrict__ nrm,
                                                    const float* __restrict__ wqkv,
                                                    __bf16* __restrict__ osT) {
  __shared__ __align__(16) __bf16 sSTN[HH][SS * CC];
  __shared__ __align__(16) __bf16 sQKV[HH][SS * 96];
  const int tid = threadIdx.x, wid = tid >> 5, lane = tid & 31;

  for (int idx = lane; idx < SS * CC; idx += 32) {
    int s = idx >> 5;
    float v = st[wid * SS * CC + idx] / (nrm[wid * SS + s] + 1e-5f);
    sSTN[wid][idx] = f2bf(v);
  }
  __syncthreads();
  for (int idx = lane; idx < SS * 96; idx += 32) {
    int s = idx / 96, j = idx % 96;
    float acc = 0.f;
#pragma unroll
    for (int c = 0; c < CC; ++c) acc += bf2f(sSTN[wid][s * CC + c]) * wqkv[c * 96 + j];
    sQKV[wid][idx] = f2bf(acc);
  }
  __syncthreads();
  for (int srow = lane; srow < SS; srow += 32) {
    float sc[SS];
    float mx = -3.0e38f;
    const __bf16* q = &sQKV[wid][srow * 96];
    for (int t = 0; t < SS; ++t) {
      const __bf16* kk = &sQKV[wid][t * 96 + 32];
      float d = 0.f;
#pragma unroll
      for (int c = 0; c < CC; ++c) d += bf2f(q[c]) * bf2f(kk[c]);
      d *= 0.17677669529663687f;  // 1/sqrt(32)
      sc[t] = d; mx = fmaxf(mx, d);
    }
    float sum = 0.f;
    for (int t = 0; t < SS; ++t) { float e = __expf(sc[t] - mx); sc[t] = e; sum += e; }
    const float rn = 1.f / sum;
    float o[CC];
#pragma unroll
    for (int c = 0; c < CC; ++c) o[c] = 0.f;
    for (int t = 0; t < SS; ++t) {
      float a = sc[t] * rn;
      const __bf16* vv = &sQKV[wid][t * 96 + 64];
#pragma unroll
      for (int c = 0; c < CC; ++c) o[c] += a * bf2f(vv[c]);
    }
    for (int c = 0; c < CC; ++c) osT[wid * CC * SS + c * SS + srow] = f2bf(o[c]);  // [h][c][s]
  }
}

// ---------------- pass3: recompute routing, scatter, proj ----------------
__global__ __launch_bounds__(256) void k_pass3(
    const float* __restrict__ x,
    const __bf16* __restrict__ wxt, const __bf16* __restrict__ wpt,
    const __bf16* __restrict__ wslt, const __bf16* __restrict__ osT,
    const float* __restrict__ bx, const float* __restrict__ bsl,
    const float* __restrict__ bp, const float* __restrict__ temp,
    float* __restrict__ out, int nTok) {
  __shared__ __align__(16) __bf16 sWsl[SS * CC];
  __shared__ __align__(16) __bf16 sOutX[16 * DD];    // [tok][d], shared by all waves
  __shared__ __align__(16) __bf16 sXm[8][16 * CC];   // per wave
  __shared__ __align__(16) __bf16 sW [8][16 * SS];   // [tok][s] per wave
  __shared__ float sBx[DD], sBp[DD], sBsl[SS], sInvT[HH];

  const int tid = threadIdx.x;
  const int wid = tid >> 5;       // head for stages 1-3, output-column group for proj
  const int lane = tid & 31;
  const int lhalf = lane >> 4;
  const int l15 = lane & 15;
  const int h = wid;

  if (tid < DD) { sBx[tid] = bx[tid]; sBp[tid] = bp[tid]; }
  if (tid < SS) sBsl[tid] = bsl[tid];
  if (tid < HH) sInvT[tid] = 1.0f / temp[tid];
  for (int i = tid; i < SS * CC; i += 256) sWsl[i] = wslt[i];
  __syncthreads();

  const int tb = blockIdx.x * 16;

  // x_mid for this head (M=16, K=256, N=32)
  v8f xm[2] = {zero8(), zero8()};
#pragma unroll
  for (int k = 0; k < 8; ++k) {
    v16bf a = afrag_f32(x + (size_t)(tb + l15) * DD, k * 32, lhalf);
#pragma unroll
    for (int nt = 0; nt < 2; ++nt) {
      v16bf b = bfrag(wxt, h * CC + nt * 16 + l15, DD, k * 32, lhalf);
      xm[nt] = wmma_bf(a, b, xm[nt]);
    }
  }
#pragma unroll
  for (int nt = 0; nt < 2; ++nt) {
    const int c = nt * 16 + l15;
    const float bb = sBx[h * CC + c];
    __bf16* dst = &sXm[wid][(8 * lhalf) * CC + c];
#pragma unroll
    for (int r = 0; r < 8; ++r) dst[r * CC] = f2bf(xm[nt][r] + bb);
  }

  // logits -> sW [tok][s] (raw)
  {
    v16bf a = bfrag(sXm[wid], l15, CC, 0, lhalf);
#pragma unroll
    for (int nt = 0; nt < 4; ++nt) {
      v16bf b = bfrag(sWsl, nt * 16 + l15, CC, 0, lhalf);
      v8f lg = wmma_bf(a, b, zero8());
      const int s = nt * 16 + l15;
      __bf16* dst = &sW[wid][(8 * lhalf) * SS + s];
#pragma unroll
      for (int r = 0; r < 8; ++r) dst[r * SS] = f2bf(lg[r]);
    }
  }

  // softmax (lanes 0..15 own the 16 tokens; rows are contiguous)
  if (lane < 16) {
    const float invT = sInvT[h];
    __bf16* row = &sW[wid][l15 * SS];
    float mx = -3.0e38f;
    for (int s = 0; s < SS; ++s) {
      float v = (bf2f(row[s]) + sBsl[s]) * invT;
      mx = fmaxf(mx, v);
    }
    float sum = 0.f;
    for (int s = 0; s < SS; ++s) {
      float v = (bf2f(row[s]) + sBsl[s]) * invT;
      float e = __expf(v - mx);
      sum += e;
      row[s] = f2bf(e);
    }
    const float rn = 1.0f / sum;
    for (int s = 0; s < SS; ++s) row[s] = f2bf(bf2f(row[s]) * rn);
  }

  // scatter: out_x_head = w @ out_slice_head (M=16, K=64, N=32); B from L2-resident osT [h][c][s]
  v8f oa[2] = {zero8(), zero8()};
#pragma unroll
  for (int ks = 0; ks < 2; ++ks) {
    v16bf a = bfrag(sW[wid], l15, SS, ks * 32, lhalf);
#pragma unroll
    for (int nt = 0; nt < 2; ++nt) {
      v16bf b = bfrag(osT + h * CC * SS, nt * 16 + l15, SS, ks * 32, lhalf);
      oa[nt] = wmma_bf(a, b, oa[nt]);
    }
  }
#pragma unroll
  for (int nt = 0; nt < 2; ++nt) {
    const int c = h * CC + nt * 16 + l15;
    __bf16* dst = &sOutX[(8 * lhalf) * DD + c];
#pragma unroll
    for (int r = 0; r < 8; ++r) dst[r * DD] = f2bf(oa[nt][r]);
  }
  __syncthreads();

  // proj: wave `wid` computes output columns [wid*32, wid*32+32) (M=16, K=256)
  v8f po[2] = {zero8(), zero8()};
#pragma unroll
  for (int k = 0; k < 8; ++k) {
    v16bf a = bfrag(sOutX, l15, DD, k * 32, lhalf);
#pragma unroll
    for (int nt = 0; nt < 2; ++nt) {
      v16bf b = bfrag(wpt, wid * CC + nt * 16 + l15, DD, k * 32, lhalf);
      po[nt] = wmma_bf(a, b, po[nt]);
    }
  }
#pragma unroll
  for (int nt = 0; nt < 2; ++nt) {
    const int col = wid * CC + nt * 16 + l15;
    const float bb = sBp[col];
#pragma unroll
    for (int r = 0; r < 8; ++r)
      out[(size_t)(tb + r + 8 * lhalf) * DD + col] = po[nt][r] + bb;
  }
}

extern "C" void kernel_launch(void* const* d_in, const int* in_sizes, int n_in,
                              void* d_out, int out_size, void* d_ws, size_t ws_size,
                              hipStream_t stream) {
  const float* x    = (const float*)d_in[0];
  const float* Wfx  = (const float*)d_in[1];
  const float* bfx  = (const float*)d_in[2];
  const float* Wx   = (const float*)d_in[3];
  const float* bx   = (const float*)d_in[4];
  const float* Wsl  = (const float*)d_in[5];
  const float* bsl  = (const float*)d_in[6];
  const float* Wqkv = (const float*)d_in[7];
  const float* Wp   = (const float*)d_in[8];
  const float* bp   = (const float*)d_in[9];
  const float* temp = (const float*)d_in[10];
  float* out = (float*)d_out;
  const int nTok = in_sizes[0] / DD;

  char* ws = (char*)d_ws;
  float*  wsSt    = (float*)(ws + 0);        // 16384 f32
  float*  wsNorm  = (float*)(ws + 65536);    // 512 f32
  __bf16* wsOsT   = (__bf16*)(ws + 67584);   // 8*32*64 bf16 (out_slice [h][c][s])
  __bf16* wsWxT   = (__bf16*)(ws + 100352);  // 256x256 bf16 transposed
  __bf16* wsWfxT  = (__bf16*)(ws + 231424);
  __bf16* wsWpT   = (__bf16*)(ws + 362496);
  __bf16* wsWslT  = (__bf16*)(ws + 493568);  // 64x32 bf16
  float*  wsPart  = (float*)(ws + 497664);   // 256 * 16896 f32 partials

  k_prep<<<776, 256, 0, stream>>>(Wfx, Wx, Wp, Wsl, wsWfxT, wsWxT, wsWpT, wsWslT);

  const int P1B = 256;
  const int iters = (nTok + 32 * P1B - 1) / (32 * P1B);
  k_pass1<<<P1B, 256, 0, stream>>>(x, wsWxT, wsWfxT, wsWslT, bx, bfx, bsl, temp,
                                   wsPart, nTok, iters);
  k_reduce<<<66, 256, 0, stream>>>(wsPart, wsSt, wsNorm, P1B);
  k_slice_attn<<<1, 256, 0, stream>>>(wsSt, wsNorm, Wqkv, wsOsT);
  k_pass3<<<nTok / 16, 256, 0, stream>>>(x, wsWxT, wsWpT, wsWslT, wsOsT,
                                         bx, bsl, bp, temp, out, nTok);
}
